// CustomMSDeformableAttention_41180146434514
// MI455X (gfx1250) — matrix operand order repaired and checked by reading, not compiled, not run on hardware
//
#include <hip/hip_runtime.h>

// ---------------- problem constants (fixed by the reference) ----------------
#define EMBED     256
#define HEADS     8
#define LEVELS    4
#define POINTS    4
#define DHEAD     32            // EMBED / HEADS == wave32 width: lane = channel
#define BS        4
#define NQ        10000
#define NUM_VALUE 19947
#define KB32      (EMBED / 32)  // 8 K-blocks of 32

typedef __attribute__((ext_vector_type(16))) _Float16 v16h;
typedef __attribute__((ext_vector_type(8)))  float    v8f;
typedef __attribute__((ext_vector_type(4)))  float    v4f;

// ===========================================================================
// Weight pre-pack: W[K=256][N] f32 -> f16 fragments in WMMA-B lane order.
// Fragment (tn, kb): lane holds column tn*16+(lane&15); half e -> K = kb*32 +
// (lane>>4)*16 + e.  Stored as 16 contiguous halves per lane (32B, coalesced).
// One wave per (tn, kb).
// ===========================================================================
__global__ __launch_bounds__(32)
void msda_pack_w_kernel(const float* __restrict__ W, _Float16* __restrict__ Bp,
                        int N) {
  const int lane = threadIdx.x;
  const int g    = lane >> 4;
  const int r    = lane & 15;
  const int tn   = blockIdx.x;          // N-tile
  const int kb   = blockIdx.y;          // K-block (32)
  const int col  = tn * 16 + r;

  const float* src = W + (size_t)(kb * 32 + g * 16) * N + col;
  v16h frag;
#pragma unroll
  for (int e = 0; e < 16; ++e) frag[e] = (_Float16)src[(size_t)e * N];

  *(v16h*)(Bp + (((size_t)tn * KB32 + kb) * 32 + lane) * 16) = frag;
}

// ===========================================================================
// WMMA GEMM: Out[M,N] = A[M,256] @ W + bias (+ Residual).
// One block = one M-tile x full N.  The 16x256 f32 A tile is converted to f16
// WMMA-A fragments and staged in LDS ONCE (8KB), then shared by all 4 waves
// (global A traffic 1x instead of N/64 x).  Each wave computes NTW N-tiles;
// K=256 fully unrolled -> NTW*8 inline v_wmma per wave.
// blockDim = (32, 4).
// ===========================================================================
template <int NTW>   // N-tiles per wave: 4 (N=256) or 2 (N=128)
__global__ __launch_bounds__(128)
void msda_gemm_wmma_kernel(const float* __restrict__ A,
                           const _Float16* __restrict__ Bp,
                           const float* __restrict__ bias,
                           float* __restrict__ Out,
                           const float* __restrict__ Residual,
                           int M, int N) {
  __shared__ _Float16 sA[KB32 * 32 * 16];       // 8KB: [kb][lane][16 halves]

  const int lane = threadIdx.x;                 // 0..31
  const int g    = lane >> 4;
  const int r    = lane & 15;
  const int wv   = threadIdx.y;                 // 0..3
  const int tm   = blockIdx.x;                  // M-tile

  int arow = tm * 16 + r;
  if (arow >= M) arow = M - 1;                  // clamp loads; stores masked
  const float* __restrict__ Arow = A + (size_t)arow * EMBED;

  // ---- stage A tile into LDS in fragment layout (each wave: 2 K-blocks) ----
#pragma unroll
  for (int i = 0; i < KB32 / 4; ++i) {
    const int kb = wv * (KB32 / 4) + i;
    // 16-bit A 16x32: one contiguous 16-float run per lane (K = g*16 + e), matching B pack.
    const float* p0 = Arow + kb * 32 + g * 16;
    const v4f a0 = *(const v4f*)(p0);
    const v4f a1 = *(const v4f*)(p0 + 4);
    const v4f a2 = *(const v4f*)(p0 + 8);
    const v4f a3 = *(const v4f*)(p0 + 12);
    v16h fa;
#pragma unroll
    for (int e = 0; e < 4; ++e) {
      fa[e]      = (_Float16)a0[e];
      fa[e + 4]  = (_Float16)a1[e];
      fa[e + 8]  = (_Float16)a2[e];
      fa[e + 12] = (_Float16)a3[e];
    }
    *(v16h*)(sA + ((size_t)kb * 32 + lane) * 16) = fa;
  }
  __syncthreads();

  // ---- compute: wave wv owns N-tiles [wv*NTW, wv*NTW+NTW) ----
  const int tbase = wv * NTW;
  const _Float16* __restrict__ Blane =
      Bp + ((size_t)tbase * KB32 * 32 + lane) * 16;

  v8f acc[NTW];
#pragma unroll
  for (int t = 0; t < NTW; ++t) acc[t] = (v8f){};

#pragma unroll
  for (int kb = 0; kb < KB32; ++kb) {
    const v16h a = *(const v16h*)(sA + ((size_t)kb * 32 + lane) * 16);
#pragma unroll
    for (int t = 0; t < NTW; ++t) {
      const v16h b = *(const v16h*)(Blane + ((size_t)(t * KB32 + kb) * 32) * 16);
      acc[t] = __builtin_amdgcn_wmma_f32_16x16x32_f16(false, a, false, b,
                                                      (short)0, acc[t],
                                                      false, false);
    }
  }

  // ---- epilogue: D lane holds column (lane&15); VGPR v holds row g*8+v ----
#pragma unroll
  for (int t = 0; t < NTW; ++t) {
    const int col = (tbase + t) * 16 + r;
    const float bn = bias[col];
#pragma unroll
    for (int v = 0; v < 8; ++v) {
      const int m = tm * 16 + g * 8 + v;
      if (m < M) {
        const size_t o = (size_t)m * N + col;
        float val = acc[t][v] + bn;
        if (Residual) val += Residual[o];
        Out[o] = val;
      }
    }
  }
}

// ===========================================================================
// Sampling kernel: one wave32 per (b, q, head); lane = channel (DHEAD == 32).
// Fuses 16-way softmax + 4-level x 4-point bilinear gather/einsum.
// Every corner gather is one coalesced 128B wave transaction.
// ===========================================================================
__device__ __forceinline__ float msda_corner(const float* __restrict__ base,
                                             int W, int H, int x, int y) {
  if ((unsigned)x < (unsigned)W && (unsigned)y < (unsigned)H)
    return base[(size_t)(y * W + x) * EMBED];
  return 0.f;
}

__global__ __launch_bounds__(256)
void msda_sample_kernel(const float* __restrict__ vproj,   // [BS][NUM_VALUE][EMBED]
                        const float* __restrict__ offs,    // [BS][NQ][H*L*P*2]
                        const float* __restrict__ logits,  // [BS][NQ][H*L*P]
                        const float* __restrict__ refpts,  // [BS][NQ][LEVELS][2]
                        float* __restrict__ mid) {         // [BS][NQ][EMBED]
  constexpr int cH[LEVELS] = {100, 50, 25, 13};
  constexpr int cW[LEVELS] = {150, 75, 38, 19};
  constexpr int cS[LEVELS] = {0, 15000, 18750, 19700};

  const int lane = threadIdx.x;                             // channel d
  const int w = blockIdx.x * blockDim.y + threadIdx.y;      // (b,q,h) id
  if (w >= BS * NQ * HEADS) return;
  const int b  = w / (NQ * HEADS);
  const int qh = w - b * (NQ * HEADS);
  const int q  = qh / HEADS;
  const int h  = qh - q * HEADS;

  const float* lp = logits + ((size_t)b * NQ + q) * (HEADS * LEVELS * POINTS)
                           + h * (LEVELS * POINTS);
  float aw[LEVELS * POINTS];
  float mx = lp[0];
#pragma unroll
  for (int i = 1; i < LEVELS * POINTS; ++i) mx = fmaxf(mx, lp[i]);
  float s = 0.f;
#pragma unroll
  for (int i = 0; i < LEVELS * POINTS; ++i) {
    const float e = __expf(lp[i] - mx);
    aw[i] = e; s += e;
  }
  const float inv = 1.f / s;

  const float* rp = refpts + ((size_t)b * NQ + q) * (LEVELS * 2);
  const float* op = offs + ((size_t)b * NQ + q) * (HEADS * LEVELS * POINTS * 2)
                         + h * (LEVELS * POINTS * 2);
  const float* vb = vproj + (size_t)b * NUM_VALUE * EMBED + h * DHEAD + lane;

  float acc = 0.f;
#pragma unroll
  for (int l = 0; l < LEVELS; ++l) {
    const int H = cH[l], Wd = cW[l];
    const float rx = rp[l * 2 + 0], ry = rp[l * 2 + 1];
    const float* base = vb + (size_t)cS[l] * EMBED;
#pragma unroll
    for (int p = 0; p < POINTS; ++p) {
      const float ox = op[(l * POINTS + p) * 2 + 0];
      const float oy = op[(l * POINTS + p) * 2 + 1];
      const float px = (rx + ox / (float)Wd) * (float)Wd - 0.5f;
      const float py = (ry + oy / (float)H)  * (float)H  - 0.5f;
      const float x0f = floorf(px), y0f = floorf(py);
      const float dx = px - x0f, dy = py - y0f;
      const int x0 = (int)x0f, y0 = (int)y0f;
      const float wgt = aw[l * POINTS + p] * inv;
      const float w00 = (1.f - dx) * (1.f - dy) * wgt;
      const float w01 = dx * (1.f - dy) * wgt;
      const float w10 = (1.f - dx) * dy * wgt;
      const float w11 = dx * dy * wgt;
      acc = fmaf(w00, msda_corner(base, Wd, H, x0,     y0    ), acc);
      acc = fmaf(w01, msda_corner(base, Wd, H, x0 + 1, y0    ), acc);
      acc = fmaf(w10, msda_corner(base, Wd, H, x0,     y0 + 1), acc);
      acc = fmaf(w11, msda_corner(base, Wd, H, x0 + 1, y0 + 1), acc);
    }
  }
  mid[((size_t)b * NQ + q) * EMBED + h * DHEAD + lane] = acc;
}

// ===========================================================================
// Launch: pack weights -> valueProj -> offProj -> attnProj -> sample ->
//         outProj(+residual). All on `stream`, deterministic, capture-safe.
// ===========================================================================
extern "C" void kernel_launch(void* const* d_in, const int* in_sizes, int n_in,
                              void* d_out, int out_size, void* d_ws, size_t ws_size,
                              hipStream_t stream) {
  (void)in_sizes; (void)n_in; (void)out_size; (void)ws_size;

  const float* query   = (const float*)d_in[0];   // [BS,NQ,EMBED]
  const float* value   = (const float*)d_in[1];   // [BS,NUM_VALUE,EMBED]
  const float* refpts  = (const float*)d_in[2];   // [BS,NQ,LEVELS,2]
  /* d_in[3] spatial_shapes (int64) - compile-time constants, unused */
  const float* w_value = (const float*)d_in[4];
  const float* b_value = (const float*)d_in[5];
  const float* w_off   = (const float*)d_in[6];
  const float* b_off   = (const float*)d_in[7];
  const float* w_attn  = (const float*)d_in[8];
  const float* b_attn  = (const float*)d_in[9];
  const float* w_out   = (const float*)d_in[10];
  const float* b_out   = (const float*)d_in[11];
  float* out = (float*)d_out;

  // ---- workspace carve-up (all fully overwritten before read) ----
  float* v_ws   = (float*)d_ws;                                   // BS*NUM_VALUE*256
  float* off_ws = v_ws   + (size_t)BS * NUM_VALUE * EMBED;        // BS*NQ*256
  float* att_ws = off_ws + (size_t)BS * NQ * EMBED;               // BS*NQ*128
  float* mid_ws = att_ws + (size_t)BS * NQ * (HEADS * LEVELS * POINTS); // BS*NQ*256
  _Float16* pw  = (_Float16*)(mid_ws + (size_t)BS * NQ * EMBED);  // packed weights
  _Float16* pw_value = pw;                       // 256x256 -> 65536 halves
  _Float16* pw_off   = pw_value + 65536;         // 256x256
  _Float16* pw_attn  = pw_off   + 65536;         // 256x128 -> 32768
  _Float16* pw_out   = pw_attn  + 32768;         // 256x256

  // ---- 0) pack the four weight matrices into f16 WMMA-B fragments ----
  msda_pack_w_kernel<<<dim3(EMBED / 16, KB32), 32, 0, stream>>>(w_value, pw_value, EMBED);
  msda_pack_w_kernel<<<dim3(EMBED / 16, KB32), 32, 0, stream>>>(w_off,   pw_off,   EMBED);
  msda_pack_w_kernel<<<dim3((HEADS * LEVELS * POINTS) / 16, KB32), 32, 0, stream>>>(w_attn, pw_attn, HEADS * LEVELS * POINTS);
  msda_pack_w_kernel<<<dim3(EMBED / 16, KB32), 32, 0, stream>>>(w_out,   pw_out,   EMBED);

  const dim3 gblk(32, 4);

  { // 1) v = value @ w_value + b_value      (M=79788, N=256)
    const int M = BS * NUM_VALUE, N = EMBED;
    msda_gemm_wmma_kernel<4><<<dim3((M + 15) / 16), gblk, 0, stream>>>(
        value, pw_value, b_value, v_ws, nullptr, M, N);
  }
  { // 2) offsets = query @ w_off + b_off    (M=40000, N=256)
    const int M = BS * NQ, N = HEADS * LEVELS * POINTS * 2;
    msda_gemm_wmma_kernel<4><<<dim3((M + 15) / 16), gblk, 0, stream>>>(
        query, pw_off, b_off, off_ws, nullptr, M, N);
  }
  { // 3) attn logits = query @ w_attn + b_attn  (M=40000, N=128)
    const int M = BS * NQ, N = HEADS * LEVELS * POINTS;
    msda_gemm_wmma_kernel<2><<<dim3((M + 15) / 16), gblk, 0, stream>>>(
        query, pw_attn, b_attn, att_ws, nullptr, M, N);
  }
  { // 4) fused softmax + bilinear sample + einsum -> mid
    dim3 blk(32, 8);
    dim3 grid((BS * NQ * HEADS + 7) / 8);
    msda_sample_kernel<<<grid, blk, 0, stream>>>(v_ws, off_ws, att_ws, refpts, mid_ws);
  }
  { // 5) out = mid @ w_out + b_out + query  (residual fused)
    const int M = BS * NQ, N = EMBED;
    msda_gemm_wmma_kernel<4><<<dim3((M + 15) / 16), gblk, 0, stream>>>(
        mid_ws, pw_out, b_out, out, query, M, N);
  }
}
